// RAFT_6038724018959
// MI455X (gfx1250) — matrix operand order repaired and checked by reading, not compiled, not run on hardware
//
#include <hip/hip_runtime.h>
#include <hip/hip_bf16.h>
#include <math.h>
#include <stdint.h>

typedef _Float16 h16;
typedef __attribute__((ext_vector_type(16))) _Float16 v16h;
typedef __attribute__((ext_vector_type(8)))  float    v8f;

#define BM  64
#define BN  128
#define BKK 32
#define LDA 34    // halfs per A-tile row (32 + pad, 4B aligned rows)
#define LDB 136   // halfs per B-tile row (128 + pad, 16B aligned rows)

// ---------------------------------------------------------------------------
// Implicit-GEMM convolution, NHWC x HWIO -> NHWC.
// A (activations) fp32 -> f16 im2col staged to LDS (vectorized when Cin%32==0)
// B (weights) pre-converted f16, staged via async global->LDS DMA (b128)
// 8x v_wmma_f32_16x16x32_f16 per K-step per wave.  act:0 none,1 relu,2 sig,3 tanh
// ---------------------------------------------------------------------------
__global__ __launch_bounds__(128) void conv_wmma_f16(
    const float* __restrict__ x, const h16* __restrict__ w,
    const float* __restrict__ bias, float* __restrict__ y,
    int Bn, int H, int W, int Cin, int Ho, int Wo, int Cout,
    int KH, int KW, int stride, int padY, int padX, int act)
{
  __shared__ __align__(16) h16 lsA[BM * LDA];
  __shared__ __align__(16) h16 lsB[BKK * LDB];

  const int tid  = threadIdx.x;
  const int wave = tid >> 5;
  const int lane = tid & 31;
  const int lm   = lane & 15;
  const int lh   = lane >> 4;

  const int mTile = blockIdx.x * BM;
  const int nTile = blockIdx.y * BN;
  const int M     = Bn * Ho * Wo;
  const int Ktot  = KH * KW * Cin;
  const bool fastA = (Cin & 31) == 0;

  // A staging: thread -> (row, 16-half chunk) of the 64x32 tile
  const int arow  = tid >> 1;
  const int akoff = (tid & 1) * 16;
  int ob = 0, oy = 0, ox = 0;
  const int gm = mTile + arow;
  const bool mvalid = gm < M;
  if (mvalid) {
    ob = gm / (Ho * Wo);
    int rem = gm - ob * (Ho * Wo);
    oy = rem / Wo;
    ox = rem - oy * Wo;
  }
  const int iy0 = oy * stride - padY;
  const int ix0 = ox * stride - padX;

  const int waveM = (wave >> 1) * 32;   // 2 waves over M
  const int waveN = (wave & 1) * 64;    // 2 waves over N

  v8f acc[2][4] = {};

  const uint64_t wbase = (uint64_t)(uintptr_t)w;

  for (int k0 = 0; k0 < Ktot; k0 += BKK) {
    // ---------------- stage A (im2col, fp32 -> f16) ----------------
    if (fastA) {
      // whole 32-K block lies inside one (fy,fx) tap; contiguous channels
      const int kf    = k0 / Cin;
      const int cbase = k0 - kf * Cin + akoff;
      const int fy = kf / KW, fx = kf - (kf / KW) * KW;
      const int iy = iy0 + fy, ix = ix0 + fx;
      h16* dstA = &lsA[arow * LDA + akoff];
      if (mvalid && (unsigned)iy < (unsigned)H && (unsigned)ix < (unsigned)W) {
        const float4* src = (const float4*)(x + ((ob * H + iy) * W + ix) * Cin + cbase);
        #pragma unroll
        for (int j = 0; j < 4; ++j) {
          const float4 v = src[j];
          dstA[4 * j + 0] = (h16)v.x; dstA[4 * j + 1] = (h16)v.y;
          dstA[4 * j + 2] = (h16)v.z; dstA[4 * j + 3] = (h16)v.w;
        }
      } else {
        #pragma unroll
        for (int j = 0; j < 16; ++j) dstA[j] = (h16)0.0f;
      }
    } else {
      #pragma unroll 4
      for (int i = 0; i < 16; ++i) {
        const int k = k0 + akoff + i;
        float v = 0.0f;
        if (mvalid && k < Ktot) {
          const int ci = k % Cin;
          const int kf = k / Cin;
          const int fx = kf % KW;
          const int fy = kf / KW;
          const int iy = iy0 + fy;
          const int ix = ix0 + fx;
          if ((unsigned)iy < (unsigned)H && (unsigned)ix < (unsigned)W)
            v = x[((ob * H + iy) * W + ix) * Cin + ci];
        }
        lsA[arow * LDA + akoff + i] = (h16)v;
      }
    }

    // ---------------- stage B (f16 weights, async DMA to LDS) ------------
    #pragma unroll
    for (int j = 0; j < 2; ++j) {
      const int c    = tid + 128 * j;      // 256 chunks of 16 halfs (32x128)
      const int brow = c >> 3;
      const int bno  = (c & 7) * 16;
      const int k    = k0 + brow;
      const int n0   = nTile + bno;
      h16* dstB = &lsB[brow * LDB + bno];
      if (k < Ktot && (n0 + 16) <= Cout) {
        const unsigned loff = (unsigned)(uintptr_t)dstB;           // LDS byte addr
        const unsigned voff = (unsigned)((k * Cout + n0) * 2);     // global byte off
        asm volatile(
            "global_load_async_to_lds_b128 %0, %1, %2\n\t"
            "global_load_async_to_lds_b128 %0, %1, %2 offset:16"
            :: "v"(loff), "v"(voff), "s"(wbase) : "memory");
      } else {
        #pragma unroll
        for (int j2 = 0; j2 < 16; ++j2)
          dstB[j2] = (k < Ktot && (n0 + j2) < Cout) ? w[k * Cout + n0 + j2]
                                                    : (h16)0.0f;
      }
    }

    // prefetch next K-step's weight chunk into cache
    if (k0 + BKK < Ktot)
      __builtin_prefetch(w + (k0 + BKK + (tid >> 3)) * Cout + nTile, 0, 1);

#if __has_builtin(__builtin_amdgcn_s_wait_asynccnt)
    __builtin_amdgcn_s_wait_asynccnt(0);
#else
    asm volatile("s_wait_asynccnt 0x0" ::: "memory");
#endif
    __syncthreads();

    // ---------------- build fragments (ISA 7.12.2 layouts) ---------------
    v16h af[2], bf[4];
    #pragma unroll
    for (int mi = 0; mi < 2; ++mi) {
      const int mr = waveM + mi * 16 + lm;
      #pragma unroll
      for (int r2 = 0; r2 < 8; ++r2) {
        const int kk = ((r2 >= 4) ? 16 : 0) + lh * 8 + (r2 & 3) * 2;
        af[mi][2 * r2]     = lsA[mr * LDA + kk];
        af[mi][2 * r2 + 1] = lsA[mr * LDA + kk + 1];
      }
    }
    #pragma unroll
    for (int ni = 0; ni < 4; ++ni) {
      const int nc = waveN + ni * 16 + lm;
      #pragma unroll
      for (int e = 0; e < 16; ++e) {
        bf[ni][e] = lsB[(((e >= 8) ? 16 : 0) + lh * 8 + (e & 7)) * LDB + nc];
      }
    }

    // ---------------- 8x v_wmma_f32_16x16x32_f16 ----------------
    #pragma unroll
    for (int mi = 0; mi < 2; ++mi)
      #pragma unroll
      for (int ni = 0; ni < 4; ++ni)
        acc[mi][ni] = __builtin_amdgcn_wmma_f32_16x16x32_f16(
            false, af[mi], false, bf[ni], (short)0, acc[mi][ni], false, false);

    __syncthreads();
  }

  // ---- epilogue: bias + activation + store (C/D layout: M = lh*8 + r) ----
  #pragma unroll
  for (int mi = 0; mi < 2; ++mi) {
    #pragma unroll
    for (int ni = 0; ni < 4; ++ni) {
      #pragma unroll
      for (int r2 = 0; r2 < 8; ++r2) {
        const int mr = mTile + waveM + mi * 16 + lh * 8 + r2;
        const int nc = nTile + waveN + ni * 16 + lm;
        if (mr < M && nc < Cout) {
          float v = acc[mi][ni][r2];
          if (bias) v += bias[nc];
          if (act == 1)      v = fmaxf(v, 0.0f);
          else if (act == 2) v = 1.0f / (1.0f + __expf(-v));
          else if (act == 3) v = tanhf(v);
          y[mr * Cout + nc] = v;
        }
      }
    }
  }
}

// ---------------------------------------------------------------------------
// fp32 -> f16 weight conversion
// ---------------------------------------------------------------------------
__global__ void cvt_f32_to_f16(const float* __restrict__ s, h16* __restrict__ d, int n)
{ int i = blockIdx.x * blockDim.x + threadIdx.x; if (i < n) d[i] = (h16)s[i]; }

// ---------------------------------------------------------------------------
// In-place instance norm (+ optional ReLU), one block per (batch, channel).
// ---------------------------------------------------------------------------
__global__ __launch_bounds__(256) void inorm_affine(
    float* __restrict__ x, const float* __restrict__ g,
    const float* __restrict__ b, int HW, int C, int relu)
{
  const int c  = blockIdx.x % C;
  const int bi = blockIdx.x / C;
  float* base = x + (size_t)bi * HW * C + c;
  float s = 0.0f, s2 = 0.0f;
  for (int i = threadIdx.x; i < HW; i += 256) {
    const float v = base[(size_t)i * C];
    s += v; s2 += v * v;
  }
  __shared__ float rs[256], rq[256];
  rs[threadIdx.x] = s; rq[threadIdx.x] = s2;
  __syncthreads();
  for (int off = 128; off > 0; off >>= 1) {
    if (threadIdx.x < off) {
      rs[threadIdx.x] += rs[threadIdx.x + off];
      rq[threadIdx.x] += rq[threadIdx.x + off];
    }
    __syncthreads();
  }
  const float mean = rs[0] / (float)HW;
  const float var  = rq[0] / (float)HW - mean * mean;
  const float inv  = rsqrtf(var + 0.001f);
  const float gg   = g[c] * inv;
  const float bb   = b[c] - mean * gg;
  for (int i = threadIdx.x; i < HW; i += 256) {
    float v = base[(size_t)i * C] * gg + bb;
    if (relu) v = fmaxf(v, 0.0f);
    base[(size_t)i * C] = v;
  }
}

// ---------------------------------------------------------------------------
// Correlation volume: one block per pixel; f1 pixel vector staged in LDS.
// ---------------------------------------------------------------------------
__global__ __launch_bounds__(128) void corr_kernel(
    const float* __restrict__ f1, const float* __restrict__ f2,
    float* __restrict__ corr, int H, int W, int C)
{
  const int p = blockIdx.x;
  const int b = p / (H * W);
  const int rr = p - b * (H * W);
  const int y = rr / W;
  const int xx = rr - y * W;
  __shared__ float sf1[256];
  for (int i = threadIdx.x; i < C; i += blockDim.x)
    sf1[i] = f1[(size_t)p * C + i];
  __syncthreads();
  for (int d = threadIdx.x; d < 81; d += blockDim.x) {
    const int dy = d / 9 - 4;
    const int dx = d % 9 - 4;
    const int yy = y + dy, xq = xx + dx;
    float acc = 0.0f;
    if ((unsigned)yy < (unsigned)H && (unsigned)xq < (unsigned)W) {
      const float* f2p = f2 + (((size_t)b * H + yy) * W + xq) * C;
      for (int ci = 0; ci < C; ++ci) acc += sf1[ci] * f2p[ci];
    }
    corr[(size_t)p * 81 + d] = acc / (float)C;
  }
}

// ---------------------------------------------------------------------------
// Elementwise helpers
// ---------------------------------------------------------------------------
__global__ void ew_tanh(const float* __restrict__ a, float* __restrict__ o, int n)
{ int i = blockIdx.x * blockDim.x + threadIdx.x; if (i < n) o[i] = tanhf(a[i]); }

__global__ void fill_zero(float* __restrict__ p, int n)
{ int i = blockIdx.x * blockDim.x + threadIdx.x; if (i < n) p[i] = 0.0f; }

__global__ void ew_mul(const float* __restrict__ a, const float* __restrict__ b,
                       float* __restrict__ o, int n)
{ int i = blockIdx.x * blockDim.x + threadIdx.x; if (i < n) o[i] = a[i] * b[i]; }

__global__ void ew_gru(float* __restrict__ h, const float* __restrict__ z,
                       const float* __restrict__ q, int n)
{ int i = blockIdx.x * blockDim.x + threadIdx.x;
  if (i < n) h[i] = (1.0f - z[i]) * h[i] + z[i] * q[i]; }

__global__ void ew_add(float* __restrict__ a, const float* __restrict__ b, int n)
{ int i = blockIdx.x * blockDim.x + threadIdx.x; if (i < n) a[i] += b[i]; }

__global__ void concat2_k(const float* __restrict__ a, int Ca,
                          const float* __restrict__ b, int Cb,
                          float* __restrict__ o, int P)
{
  const int C = Ca + Cb;
  const int tot = P * C;
  int i = blockIdx.x * blockDim.x + threadIdx.x;
  if (i >= tot) return;
  const int p = i / C;
  const int c = i - p * C;
  o[i] = (c < Ca) ? a[p * Ca + c] : b[p * Cb + (c - Ca)];
}

// ---------------------------------------------------------------------------
// Half-pixel bilinear upsample of flow (h,w) -> (TH,TW), scaled by (TW/w,TH/h)
// ---------------------------------------------------------------------------
__global__ void upsample_flow_k(const float* __restrict__ flow, float* __restrict__ out,
                                int Bn, int h, int w, int TH, int TW)
{
  const int total = Bn * TH * TW * 2;
  int idx = blockIdx.x * blockDim.x + threadIdx.x;
  if (idx >= total) return;
  const int c = idx & 1;
  int t = idx >> 1;
  const int xo = t % TW; t /= TW;
  const int yo = t % TH;
  const int b  = t / TH;
  const float sy = (float)h / (float)TH;
  const float sx = (float)w / (float)TW;
  const float fy = ((float)yo + 0.5f) * sy - 0.5f;
  const float fx = ((float)xo + 0.5f) * sx - 0.5f;
  const float y0f = floorf(fy), x0f = floorf(fx);
  const float wy = fy - y0f, wx = fx - x0f;
  int y0 = (int)y0f, x0 = (int)x0f;
  int y1 = y0 + 1, x1 = x0 + 1;
  y0 = min(max(y0, 0), h - 1); y1 = min(max(y1, 0), h - 1);
  x0 = min(max(x0, 0), w - 1); x1 = min(max(x1, 0), w - 1);
  const float* base = flow + ((size_t)b * h * w) * 2 + c;
  const float v00 = base[(y0 * w + x0) * 2];
  const float v01 = base[(y0 * w + x1) * 2];
  const float v10 = base[(y1 * w + x0) * 2];
  const float v11 = base[(y1 * w + x1) * 2];
  const float v = v00 * (1.f - wy) * (1.f - wx) + v01 * (1.f - wy) * wx
                + v10 * wy * (1.f - wx)        + v11 * wy * wx;
  const float scale = (c == 0) ? (float)TW / (float)w : (float)TH / (float)h;
  out[idx] = v * scale;
}

// ---------------------------------------------------------------------------
// Host orchestration
// ---------------------------------------------------------------------------
extern "C" void kernel_launch(void* const* d_in, const int* in_sizes, int n_in,
                              void* d_out, int out_size, void* d_ws, size_t ws_size,
                              hipStream_t stream)
{
  (void)in_sizes; (void)n_in; (void)out_size; (void)ws_size;
  auto in = [&](int i) -> const float* { return (const float*)d_in[i]; };
  float* ws  = (float*)d_ws;
  float* out = (float*)d_out;

  const int B = 2, H0 = 384, W0 = 512;
  const int H1 = 192, W1 = 256, H2 = 96, W2 = 128, H3 = 48, W3 = 64;

  // ---- workspace layout (float offsets) ----
  const size_t oA = 0, oB2 = 6291456, oC = 12582912, oD = 14942208,
               oE = 17301504, oF = 18087936;
  const size_t oF1 = 18874368, oF2 = 20447232, oCTX = 22020096,
               oH = 22806528, oCORR = 23592960, oFLOW = 24090624;
  const size_t oWH = 24102912;   // f16 weights region (h16*)
  // update-block scratch aliases the (finished) encoder region
  const size_t oCAT = 0, oM1 = 509952, oM2 = 1296384, oM = 1689600,
               oXH = 1886208, oZ = 2869248, oR = 3655680, oRH = 4442112,
               oXQ = 5228544, oQ = 6211584, oFH1 = 6998016, oDF = 8570880;

  float *bufA = ws + oA,  *bufB = ws + oB2, *bufC = ws + oC;
  float *bufD = ws + oD,  *bufE = ws + oE,  *bufF = ws + oF;
  float *f1 = ws + oF1, *f2 = ws + oF2, *ctx = ws + oCTX;
  float *hb = ws + oH, *corr = ws + oCORR, *flow = ws + oFLOW;

  auto ew = [&](int n) { return dim3((n + 255) / 256); };

  // ---- convert all conv weights to f16 once (deterministic every call) ----
  h16* whbase = (h16*)(ws + oWH);
  const h16* wp[64] = {};
  size_t hoff = 0;
  auto cvtw = [&](int idx, int n) {
    h16* dst = whbase + hoff;
    hoff += ((size_t)n + 7) & ~(size_t)7;   // keep 16B alignment
    cvt_f32_to_f16<<<ew(n), 256, 0, stream>>>(in(idx), dst, n);
    wp[idx] = dst;
  };
  for (int e = 0; e < 2; ++e) {
    const int base = (e == 0) ? 2 : 25;          // cenc, fenc
    const int cOut = (e == 0) ? 128 : 256;
    cvtw(base + 1, 7 * 7 * 3 * 64);
    cvtw(base + 6, 3 * 3 * 64 * 64);
    cvtw(base + 7, 3 * 3 * 64 * 64);
    cvtw(base + 12, 3 * 3 * 64 * 96);
    cvtw(base + 13, 3 * 3 * 96 * 96);
    cvtw(base + 18, 3 * 3 * 96 * 128);
    cvtw(base + 19, 3 * 3 * 128 * 128);
    cvtw(base + 22, 128 * cOut);
  }
  cvtw(50, 3 * 3 * 128 * 256);   // fh_w1
  cvtw(51, 3 * 3 * 256 * 2);     // fh_w2
  cvtw(53, 3 * 3 * 160 * 128);   // gq_w
  cvtw(55, 3 * 3 * 160 * 128);   // gr_w
  cvtw(57, 3 * 3 * 160 * 128);   // gz_w
  cvtw(61, 1 * 1 * 83 * 128);    // me_w1
  cvtw(62, 3 * 3 * 128 * 64);    // me_w2
  cvtw(63, 3 * 3 * 64 * 32);     // me_w3

  auto conv = [&](const float* xx, int widx, const float* bb, float* yy,
                  int Bn, int Hh, int Ww, int Ci, int Ho, int Wo, int Co,
                  int KH, int KW, int s, int act) {
    int padY = (Ho - 1) * s + KH - Hh; padY = padY > 0 ? padY / 2 : 0;
    int padX = (Wo - 1) * s + KW - Ww; padX = padX > 0 ? padX / 2 : 0;
    const int M = Bn * Ho * Wo;
    dim3 grid((M + BM - 1) / BM, (Co + BN - 1) / BN);
    conv_wmma_f16<<<grid, 128, 0, stream>>>(xx, wp[widx], bb, yy, Bn, Hh, Ww, Ci,
                                            Ho, Wo, Co, KH, KW, s, padY, padX, act);
  };
  auto inorm = [&](float* xx, const float* g, const float* b, int Bn, int HW,
                   int C, int relu) {
    inorm_affine<<<dim3(Bn * C), 256, 0, stream>>>(xx, g, b, HW, C, relu);
  };

  // Encoder leaves (alphabetical pytree order):
  // +0 bout, +1 conv1, +2..7 l1{b1,b2,g1,g2,w1,w2}, +8..13 l2{...},
  // +14..19 l3{...}, +20 n1b, +21 n1g, +22 wout
  auto run_enc = [&](const float* img, int base, float* dest, int cOut) {
    conv(img, base + 1, nullptr, bufA, B, H0, W0, 3, H1, W1, 64, 7, 7, 2, 0);
    inorm(bufA, in(base + 21), in(base + 20), B, H1 * W1, 64, 1);
    conv(bufA, base + 6, nullptr, bufB, B, H1, W1, 64, H1, W1, 64, 3, 3, 1, 0);
    inorm(bufB, in(base + 4), in(base + 2), B, H1 * W1, 64, 1);
    conv(bufB, base + 7, nullptr, bufA, B, H1, W1, 64, H1, W1, 64, 3, 3, 1, 0);
    inorm(bufA, in(base + 5), in(base + 3), B, H1 * W1, 64, 1);
    conv(bufA, base + 12, nullptr, bufC, B, H1, W1, 64, H2, W2, 96, 3, 3, 2, 0);
    inorm(bufC, in(base + 10), in(base + 8), B, H2 * W2, 96, 1);
    conv(bufC, base + 13, nullptr, bufD, B, H2, W2, 96, H2, W2, 96, 3, 3, 1, 0);
    inorm(bufD, in(base + 11), in(base + 9), B, H2 * W2, 96, 1);
    conv(bufD, base + 18, nullptr, bufE, B, H2, W2, 96, H3, W3, 128, 3, 3, 2, 0);
    inorm(bufE, in(base + 16), in(base + 14), B, H3 * W3, 128, 1);
    conv(bufE, base + 19, nullptr, bufF, B, H3, W3, 128, H3, W3, 128, 3, 3, 1, 0);
    inorm(bufF, in(base + 17), in(base + 15), B, H3 * W3, 128, 1);
    conv(bufF, base + 22, in(base + 0), dest, B, H3, W3, 128, H3, W3, cOut, 1, 1, 1, 0);
  };

  // pytree order: image1(0), image2(1), cenc(2..24), fenc(25..47), upd(48..63)
  run_enc(in(0), 25, f1, 256);
  run_enc(in(1), 25, f2, 256);
  run_enc(in(0), 2, ctx, 128);

  const int n3 = B * H3 * W3;
  ew_tanh<<<ew(n3 * 128), 256, 0, stream>>>(ctx, hb, n3 * 128);   // net = tanh(ctx)
  corr_kernel<<<dim3(n3), 128, 0, stream>>>(f1, f2, corr, H3, W3, 256);
  fill_zero<<<ew(n3 * 2), 256, 0, stream>>>(flow, n3 * 2);

  float *cat83 = ws + oCAT, *m1 = ws + oM1, *m2 = ws + oM2, *mm = ws + oM;
  float *xh = ws + oXH, *zb = ws + oZ, *rb = ws + oR, *rh = ws + oRH;
  float *xq = ws + oXQ, *qb = ws + oQ, *fh1 = ws + oFH1, *df = ws + oDF;

  // upd leaves: 48 fh_b1, 49 fh_b2, 50 fh_w1, 51 fh_w2, 52 gq_b, 53 gq_w,
  //             54 gr_b, 55 gr_w, 56 gz_b, 57 gz_w, 58 me_b1, 59 me_b2,
  //             60 me_b3, 61 me_w1, 62 me_w2, 63 me_w3
  for (int it = 0; it < 8; ++it) {
    concat2_k<<<ew(n3 * 83), 256, 0, stream>>>(corr, 81, flow, 2, cat83, n3);
    conv(cat83, 61, in(58), m1, B, H3, W3, 83,  H3, W3, 128, 1, 1, 1, 1);
    conv(m1,    62, in(59), m2, B, H3, W3, 128, H3, W3, 64,  3, 3, 1, 1);
    conv(m2,    63, in(60), mm, B, H3, W3, 64,  H3, W3, 32,  3, 3, 1, 1);
    concat2_k<<<ew(n3 * 160), 256, 0, stream>>>(mm, 32, hb, 128, xh, n3);
    conv(xh, 57, in(56), zb, B, H3, W3, 160, H3, W3, 128, 3, 3, 1, 2);  // z
    conv(xh, 55, in(54), rb, B, H3, W3, 160, H3, W3, 128, 3, 3, 1, 2);  // r
    ew_mul<<<ew(n3 * 128), 256, 0, stream>>>(rb, hb, rh, n3 * 128);
    concat2_k<<<ew(n3 * 160), 256, 0, stream>>>(mm, 32, rh, 128, xq, n3);
    conv(xq, 53, in(52), qb, B, H3, W3, 160, H3, W3, 128, 3, 3, 1, 3);  // q
    ew_gru<<<ew(n3 * 128), 256, 0, stream>>>(hb, zb, qb, n3 * 128);
    conv(hb,  50, in(48), fh1, B, H3, W3, 128, H3, W3, 256, 3, 3, 1, 1);
    conv(fh1, 51, in(49), df,  B, H3, W3, 256, H3, W3, 2,   3, 3, 1, 0);
    ew_add<<<ew(n3 * 2), 256, 0, stream>>>(flow, df, n3 * 2);
    upsample_flow_k<<<ew(B * H0 * W0 * 2), 256, 0, stream>>>(
        flow, out + (size_t)it * B * H0 * W0 * 2, B, H3, W3, H0, W0);
  }
}
